// SelfAttention_41334765257086
// MI455X (gfx1250) — matrix operand order repaired and checked by reading, not compiled, not run on hardware
//
#include <hip/hip_runtime.h>
#include <hip/hip_bf16.h>

typedef __attribute__((ext_vector_type(16))) _Float16 v16h;
typedef __attribute__((ext_vector_type(8)))  _Float16 v8h;
typedef __attribute__((ext_vector_type(8)))  float    v8f;

#define B_  4
#define S_  2048
#define D_  1024
#define H_  16
#define HD_ 64
#define LDK 72   // padded LDS stride (halves) for 64-half rows -> bank-conflict free
#define LDW 40   // padded LDS stride (halves) for 32-half rows

static __device__ __forceinline__ int laneid() { return threadIdx.x & 31; }

// D = A*B + C, 16x16 f32 accum, f16 inputs, K=32
static __device__ __forceinline__ v8f wmma_f16(v16h a, v16h b, v8f c) {
  return __builtin_amdgcn_wmma_f32_16x16x32_f16(false, a, false, b, (short)0, c,
                                                false, false);
}

// Async global->LDS copy: each lane moves 16 bytes. ldsaddr = low 32 bits of
// the generic pointer (== LDS byte offset), gaddr = 64-bit global address.
static __device__ __forceinline__ void async_copy16(unsigned ldsaddr,
                                                    unsigned long long gaddr) {
  asm volatile("global_load_async_to_lds_b128 %0, %1, off"
               :: "v"(ldsaddr), "v"(gaddr)
               : "memory");
}
static __device__ __forceinline__ void wait_async_le(int n) {
  switch (n) {
    case 0:  asm volatile("s_wait_asynccnt 0x0" ::: "memory"); break;
    case 1:  asm volatile("s_wait_asynccnt 0x1" ::: "memory"); break;
    case 4:  asm volatile("s_wait_asynccnt 0x4" ::: "memory"); break;
    default: asm volatile("s_wait_asynccnt 0x0" ::: "memory"); break;
  }
}

// ---- A operand (16xK tile, row-major source, leading dim ld, chunk k0) ----
// Lane 0-15 row M=lane, halves {k0+0..7, k0+16..23};
// lane 16-31 row M=lane-16, halves {k0+8..15, k0+24..31}.
static __device__ __forceinline__ v16h loadA_f16(const _Float16* base, int ld, int k0) {
  const int L = laneid();
  const _Float16* p = base + (L & 15) * ld + k0 + ((L >> 4) << 3);
  v8h lo = *(const v8h*)p;
  v8h hi = *(const v8h*)(p + 16);
  return __builtin_shufflevector(lo, hi, 0, 1, 2, 3, 4, 5, 6, 7,
                                 8, 9, 10, 11, 12, 13, 14, 15);
}

static __device__ __forceinline__ v16h loadA_f32(const float* base, int ld, int k0) {
  const int L = laneid();
  const float* p = base + (L & 15) * ld + k0 + ((L >> 4) << 3);
  v16h a;
#pragma unroll
  for (int i = 0; i < 8; ++i) {
    a[i]     = (_Float16)p[i];
    a[8 + i] = (_Float16)p[16 + i];
  }
  return a;
}

// ---- B operand (Kx16 tile). Source M row-major with rows = output columns:
// element(kr, n) = M[n0+n][k0+kr]. Lane 0-15 col N=lane rows {k0+0..7, k0+16..23};
// lane 16-31 col N=lane-16 rows {k0+8..15, k0+24..31}.
static __device__ __forceinline__ v16h loadB_f16(const _Float16* M, int ld, int n0, int k0) {
  const int L = laneid();
  const _Float16* p = M + (n0 + (L & 15)) * ld + k0 + ((L >> 4) << 3);
  v8h lo = *(const v8h*)p;
  v8h hi = *(const v8h*)(p + 16);
  return __builtin_shufflevector(lo, hi, 0, 1, 2, 3, 4, 5, 6, 7,
                                 8, 9, 10, 11, 12, 13, 14, 15);
}

static __device__ __forceinline__ v16h loadB_f32(const float* M, int ld, int n0, int k0) {
  const int L = laneid();
  const float* p = M + (n0 + (L & 15)) * ld + k0 + ((L >> 4) << 3);
  v16h b;
#pragma unroll
  for (int i = 0; i < 16; ++i) b[i] = (_Float16)p[i < 8 ? i : i + 8];
  return b;
}

// ------------------------------------------------------------------
// Kernel 1: fp32 -> f16 convert (for Wo)
// ------------------------------------------------------------------
__global__ __launch_bounds__(256) void k_convert(const float* __restrict__ src,
                                                 _Float16* __restrict__ dst, int n) {
  int i = blockIdx.x * blockDim.x + threadIdx.x;
  if (i < n) dst[i] = (_Float16)src[i];
}

// ------------------------------------------------------------------
// Kernel 2: per-head QKV projection. y = x @ W.T per head.
// Writes q,k as [b,h,s,d] f16 and v TRANSPOSED as [b,h,d,s] f16.
// One wave per 16-row tile of one head.
// ------------------------------------------------------------------
__global__ __launch_bounds__(256) void k_proj(
    const float* __restrict__ xv, const float* __restrict__ xk,
    const float* __restrict__ xq, const float* __restrict__ Wv,
    const float* __restrict__ Wk, const float* __restrict__ Wq,
    _Float16* __restrict__ qf, _Float16* __restrict__ kf,
    _Float16* __restrict__ vt) {
  const int wid = blockIdx.x * (blockDim.x >> 5) + (threadIdx.x >> 5);
  const int ST = S_ / 16;
  const int s0 = (wid % ST) * 16;
  const int h  = (wid / ST) % H_;
  const int b  = wid / (ST * H_);
  const int L  = threadIdx.x & 31;

  const float* srcs[3] = {xq, xk, xv};
  const float* Ws[3]   = {Wq, Wk, Wv};

#pragma unroll
  for (int t = 0; t < 3; ++t) {
    const float* xb = srcs[t] + (b * S_ + s0) * D_ + h * HD_;
    v16h a0 = loadA_f32(xb, D_, 0);
    v16h a1 = loadA_f32(xb, D_, 32);
#pragma unroll
    for (int e = 0; e < 4; ++e) {
      v8f acc = {};
      acc = wmma_f16(a0, loadB_f32(Ws[t], HD_, e * 16, 0), acc);
      acc = wmma_f16(a1, loadB_f32(Ws[t], HD_, e * 16, 32), acc);
      v8h pk;
#pragma unroll
      for (int g = 0; g < 8; ++g) pk[g] = (_Float16)acc[g];
      if (t < 2) {
        _Float16* dst = (t == 0 ? qf : kf) +
                        ((b * H_ + h) * S_ + s0) * HD_ + e * 16 + (L & 15);
#pragma unroll
        for (int g = 0; g < 8; ++g)
          dst[(g + ((L >> 4) << 3)) * HD_] = pk[g];
      } else {
        _Float16* dst = vt + ((b * H_ + h) * HD_ + e * 16 + (L & 15)) * S_ +
                        s0 + ((L >> 4) << 3);
        *(v8h*)dst = pk;
      }
    }
  }
}

// ------------------------------------------------------------------
// Kernel 3: flash attention. Block = 8 waves sharing (b,h), 128 queries.
// 64-key steps; K/V tiles double-buffered in LDS via async global->LDS DMA
// (bank-conflict-free padded stride). Per step and wave: 8 WMMA (QK^T),
// online softmax with WMMA row-sum (P x ones), 8 WMMA (P*V).
// ------------------------------------------------------------------
__global__ __launch_bounds__(256) void k_attn(const _Float16* __restrict__ qf,
                                              const _Float16* __restrict__ kf,
                                              const _Float16* __restrict__ vt,
                                              _Float16* __restrict__ att) {
  __shared__ __align__(16) _Float16 Kls[2][64 * LDK];  // [key][d], padded
  __shared__ __align__(16) _Float16 Vls[2][64 * LDK];  // [d][key], padded
  __shared__ __align__(16) _Float16 Pl[8][16 * LDK];   // per-wave P, padded
  const int tid = threadIdx.x;
  const int w   = tid >> 5;
  const int L   = tid & 31;
  const int QB  = S_ / 128;  // 16 query-blocks per (b,h)
  const int qb  = (blockIdx.x % QB) * 128;
  const int h   = (blockIdx.x / QB) % H_;
  const int b   = blockIdx.x / (QB * H_);
  const int q0  = qb + w * 16;

  const _Float16* qh = qf + ((size_t)((b * H_ + h) * S_) + q0) * HD_;
  const _Float16* kh = kf + (size_t)((b * H_ + h) * S_) * HD_;
  const _Float16* vh = vt + (size_t)((b * H_ + h) * HD_) * S_;

  // One 8KB tile = 512 lane-transfers of 16B = 2 block-wide instructions.
  auto issue = [&](int kk, int bi) {
#pragma unroll
    for (int p = 0; p < 2; ++p) {
      const int row = p * 32 + (tid >> 3);  // 0..63
      const int seg = tid & 7;              // 8 x 16B per 128B row
      unsigned long long gk =
          (unsigned long long)(uintptr_t)(kh + (size_t)(kk + row) * HD_) +
          (unsigned)(seg * 16);
      unsigned lk = (unsigned)(uintptr_t)(&Kls[bi][0]) +
                    (unsigned)(row * (LDK * 2) + seg * 16);
      async_copy16(lk, gk);
      unsigned long long gv =
          (unsigned long long)(uintptr_t)(vh + (size_t)row * S_ + kk) +
          (unsigned)(seg * 16);
      unsigned lv = (unsigned)(uintptr_t)(&Vls[bi][0]) +
                    (unsigned)(row * (LDK * 2) + seg * 16);
      async_copy16(lv, gv);
    }
  };

  v16h aq0 = loadA_f16(qh, HD_, 0);
  v16h aq1 = loadA_f16(qh, HD_, 32);

  v16h onesB;
#pragma unroll
  for (int i = 0; i < 16; ++i) onesB[i] = (_Float16)1.0f;

  v8f O[4] = {};
  float mrow[8], lrow[8];
#pragma unroll
  for (int g = 0; g < 8; ++g) { mrow[g] = -1e30f; lrow[g] = 0.f; }

  issue(0, 0);
  int cur = 0;
  for (int kk = 0; kk < S_; kk += 64) {
    const bool more = (kk + 64) < S_;
    if (more) {
      issue(kk + 64, cur ^ 1);
      wait_async_le(4);  // the 4 ops for `cur` have completed
    } else {
      wait_async_le(0);
    }
    __syncthreads();

    const _Float16* Kb = &Kls[cur][0];
    const _Float16* Vb = &Vls[cur][0];
    _Float16* Pw = &Pl[w][0];

    v8f s4[4];
#pragma unroll
    for (int t = 0; t < 4; ++t) {
      v8f acc = {};
      acc = wmma_f16(aq0, loadB_f16(Kb, LDK, t * 16, 0), acc);
      acc = wmma_f16(aq1, loadB_f16(Kb, LDK, t * 16, 32), acc);
      s4[t] = acc;
    }

    const float sc = 0.03125f;  // 1/sqrt(D) = 1/32
    float lscale[8];
#pragma unroll
    for (int g = 0; g < 8; ++g) {
      float x0 = s4[0][g] * sc, x1 = s4[1][g] * sc;
      float x2 = s4[2][g] * sc, x3 = s4[3][g] * sc;
      float tmax = fmaxf(fmaxf(x0, x1), fmaxf(x2, x3));
#pragma unroll
      for (int m = 1; m < 16; m <<= 1)
        tmax = fmaxf(tmax, __shfl_xor(tmax, m, 32));
      float mnew = fmaxf(mrow[g], tmax);
      lscale[g]  = __expf(mrow[g] - mnew);
      mrow[g]    = mnew;
      O[0][g] *= lscale[g]; O[1][g] *= lscale[g];
      O[2][g] *= lscale[g]; O[3][g] *= lscale[g];
      const int r = (g + ((L >> 4) << 3)) * LDK + (L & 15);
      Pw[r]      = (_Float16)__expf(x0 - mnew);
      Pw[r + 16] = (_Float16)__expf(x1 - mnew);
      Pw[r + 32] = (_Float16)__expf(x2 - mnew);
      Pw[r + 48] = (_Float16)__expf(x3 - mnew);
    }
    // wave-private LDS; DS ops from one wave execute in order.
    __builtin_amdgcn_wave_barrier();
    asm volatile("" ::: "memory");

    v16h ap0 = loadA_f16(Pw, LDK, 0);
    v16h ap1 = loadA_f16(Pw, LDK, 32);

    // row-sum on the matrix unit: rs = P x ones (all 16 cols = row sum)
    v8f rs = {};
    rs = wmma_f16(ap0, onesB, rs);
    rs = wmma_f16(ap1, onesB, rs);
#pragma unroll
    for (int g = 0; g < 8; ++g) lrow[g] = lrow[g] * lscale[g] + rs[g];

#pragma unroll
    for (int t = 0; t < 4; ++t) {
      O[t] = wmma_f16(ap0, loadB_f16(Vb, LDK, t * 16, 0), O[t]);
      O[t] = wmma_f16(ap1, loadB_f16(Vb, LDK, t * 16, 32), O[t]);
    }

    __syncthreads();  // all waves done reading `cur` before it is refilled
    cur ^= 1;
  }

  _Float16* ao = att + (b * S_ + q0) * D_ + h * HD_;
#pragma unroll
  for (int g = 0; g < 8; ++g) {
    float inv = 1.0f / lrow[g];
    const int r = g + ((L >> 4) << 3);
#pragma unroll
    for (int t = 0; t < 4; ++t)
      ao[r * D_ + t * 16 + (L & 15)] = (_Float16)(O[t][g] * inv);
  }
}

// ------------------------------------------------------------------
// Kernel 4: output projection out = att @ Wo.T + bo (8192x1024x1024).
// Block = 8 waves sharing one 64-column Wo panel, double-buffered in LDS
// via async DMA (padded stride). One wave = 16x64 tile; 128 WMMAs.
// ------------------------------------------------------------------
__global__ __launch_bounds__(256) void k_outproj(const _Float16* __restrict__ att,
                                                 const _Float16* __restrict__ wo,
                                                 const float* __restrict__ bo,
                                                 float* __restrict__ out) {
  __shared__ __align__(16) _Float16 Wls[2][64 * LDW];  // [n][k] panel, padded
  const int tid = threadIdx.x;
  const int w   = tid >> 5;
  const int L   = tid & 31;
  const int nb  = blockIdx.x & 15;   // 16 column panels
  const int mg  = blockIdx.x >> 4;   // 64 row groups (8 m-tiles each)
  const int n0  = nb * 64;
  const int m0  = (mg * 8 + w) * 16;
  const _Float16* arow = att + (size_t)m0 * D_;

  auto issueW = [&](int kc, int bi) {
    const int row = tid >> 2, seg = tid & 3;  // 64 rows x 64B
    unsigned long long gw =
        (unsigned long long)(uintptr_t)(wo + (size_t)(n0 + row) * D_ + kc) +
        (unsigned)(seg * 16);
    unsigned lw = (unsigned)(uintptr_t)(&Wls[bi][0]) +
                  (unsigned)(row * (LDW * 2) + seg * 16);
    async_copy16(lw, gw);
  };

  v8f acc[4] = {};
  issueW(0, 0);
  int cur = 0;
  for (int kc = 0; kc < D_; kc += 32) {
    const bool more = (kc + 32) < D_;
    if (more) {
      issueW(kc + 32, cur ^ 1);
      wait_async_le(1);
    } else {
      wait_async_le(0);
    }
    __syncthreads();

    v16h a = loadA_f16(arow, D_, kc);
#pragma unroll
    for (int t = 0; t < 4; ++t)
      acc[t] = wmma_f16(a, loadB_f16(&Wls[cur][0], LDW, t * 16, 0), acc[t]);

    __syncthreads();
    cur ^= 1;
  }
#pragma unroll
  for (int t = 0; t < 4; ++t) {
    const int n = n0 + t * 16 + (L & 15);
    const float bias = bo[n];
#pragma unroll
    for (int g = 0; g < 8; ++g) {
      const int m = m0 + g + ((L >> 4) << 3);
      out[(size_t)m * D_ + n] = acc[t][g] + bias;
    }
  }
}

// ------------------------------------------------------------------
extern "C" void kernel_launch(void* const* d_in, const int* in_sizes, int n_in,
                              void* d_out, int out_size, void* d_ws,
                              size_t ws_size, hipStream_t stream) {
  const float* xv = (const float*)d_in[0];
  const float* xk = (const float*)d_in[1];
  const float* xq = (const float*)d_in[2];
  const float* Wv = (const float*)d_in[3];
  const float* Wk = (const float*)d_in[4];
  const float* Wq = (const float*)d_in[5];
  const float* Wo = (const float*)d_in[6];
  const float* bo = (const float*)d_in[7];
  float* out = (float*)d_out;

  const size_t BSD = (size_t)B_ * S_ * D_;  // 8,388,608
  _Float16* qf   = (_Float16*)d_ws;
  _Float16* kf   = qf + BSD;
  _Float16* vt   = kf + BSD;
  _Float16* att  = vt + BSD;
  _Float16* wo16 = att + BSD;  // ~66 MB workspace total

  k_convert<<<(D_ * D_ + 255) / 256, 256, 0, stream>>>(Wo, wo16, D_ * D_);

  const int waves = B_ * H_ * (S_ / 16);  // 8192
  k_proj<<<waves / 8, 256, 0, stream>>>(xv, xk, xq, Wv, Wk, Wq, qf, kf, vt);

  const int ablocks = B_ * H_ * (S_ / 128);  // 1024 blocks, 8 waves each
  k_attn<<<ablocks, 256, 0, stream>>>(qf, kf, vt, att);

  const int gblocks = ((B_ * S_ / 16) / 8) * (D_ / 64);  // 1024 blocks
  k_outproj<<<gblocks, 256, 0, stream>>>(att, wo16, bo, out);
}
